// SAMBLOCK_15625091023390
// MI455X (gfx1250) — matrix operand order repaired and checked by reading, not compiled, not run on hardware
//
#include <hip/hip_runtime.h>
#include <hip/hip_bf16.h>

typedef __attribute__((ext_vector_type(16))) _Float16 v16h;
typedef __attribute__((ext_vector_type(8)))  float    v8f;

// Problem constants (B=16, H=W=112, C=128, heads=4, hd=32, ws=7)
#define MTOK   200704      // 16*112*112 rows, multiple of 128
#define CDIM   128
#define NWIN   4096        // 16 * 16*16 windows
#define ATTN_SCALE 0.17677669529663687f

// ---- row index remapping: token order <-> window-partitioned order ----
__device__ __forceinline__ int tok_to_win_row(int t) {
    int b  = t / 12544;  int rem = t % 12544;
    int y  = rem / 112;  int x   = rem % 112;
    int wy = y / 7, iy = y % 7, wx = x / 7, ix = x % 7;
    return ((b * 16 + wy) * 16 + wx) * 49 + iy * 7 + ix;
}
__device__ __forceinline__ int win_to_tok_row(int wr) {
    int win = wr / 49, n = wr % 49;
    int b = win / 256; int r = win % 256;
    int wy = r / 16, wx = r % 16, iy = n / 7, ix = n % 7;
    return b * 12544 + (wy * 7 + iy) * 112 + wx * 7 + ix;
}

// A-fragment (16x32 f16) K-index for element e, half-select hs
__device__ __forceinline__ int afrag_k(int e, int hs) {
    return (hs << 4) + e;
}

// =================== LayerNorm (f32 in -> f16 out) =====================
template<bool REMAP>
__global__ __launch_bounds__(256) void ln_kernel(
    const float* __restrict__ X, const float* __restrict__ gam,
    const float* __restrict__ bet, _Float16* __restrict__ out, int Mrows)
{
    int tid = threadIdx.x, lane = tid & 31, w = tid >> 5;
    int row = blockIdx.x * 8 + w;
    if (row >= Mrows) return;
    const float* xr = X + (size_t)row * CDIM;
    float val[4];
    float s = 0.f;
#pragma unroll
    for (int j = 0; j < 4; ++j) { val[j] = xr[lane * 4 + j]; s += val[j]; }
#pragma unroll
    for (int off = 1; off < 32; off <<= 1) s += __shfl_xor(s, off, 32);
    float mu = s * (1.0f / 128.0f);
    float vs = 0.f;
#pragma unroll
    for (int j = 0; j < 4; ++j) { float d = val[j] - mu; vs += d * d; }
#pragma unroll
    for (int off = 1; off < 32; off <<= 1) vs += __shfl_xor(vs, off, 32);
    float rstd = rsqrtf(vs * (1.0f / 128.0f) + 1e-5f);
    int orow = REMAP ? tok_to_win_row(row) : row;
    _Float16* o = out + (size_t)orow * CDIM;
#pragma unroll
    for (int j = 0; j < 4; ++j) {
        int c = lane * 4 + j;
        o[c] = (_Float16)((val[j] - mu) * rstd * gam[c] + bet[c]);
    }
}

// ---------- shared epilogue for both GEMM kernels ----------
template<int MODE>
__device__ __forceinline__ void gemm_epilogue(
    const v8f acc[2][2], int m0, int n0, int wm, int wn, int hs, int l15,
    const float* __restrict__ bias, float scale,
    _Float16* __restrict__ outH, float* __restrict__ outF,
    const float* __restrict__ resid, int Nout)
{
#pragma unroll
    for (int mt = 0; mt < 2; ++mt)
#pragma unroll
        for (int nt = 0; nt < 2; ++nt)
#pragma unroll
            for (int r = 0; r < 8; ++r) {
                int gm = m0 + wm * 32 + mt * 16 + r + 8 * hs;
                int gc = n0 + wn * 32 + nt * 16 + l15;
                float v = (acc[mt][nt][r] + bias[gc]) * scale;
                if (MODE == 1) v = 0.5f * v * (1.0f + erff(v * 0.70710678f));
                if (MODE == 0 || MODE == 1) {
                    outH[(size_t)gm * Nout + gc] = (_Float16)v;
                } else if (MODE == 2) {
                    int t = win_to_tok_row(gm);
                    outF[(size_t)t * Nout + gc] = v + resid[(size_t)t * Nout + gc];
                } else {
                    outF[(size_t)gm * Nout + gc] = v + resid[(size_t)gm * Nout + gc];
                }
            }
}

// ======== WMMA GEMM, K==128: full A block staged once, N-loop inside ========
// A: M x 128 (f16 row-major), W: Nout x 128 (f32 row-major)
// MODE 0: f16 out; 1: f16+GELU; 2: f32 + win->tok remap + residual; 3: f32 + residual
template<int MODE>
__global__ __launch_bounds__(256) void gemm_wmma_nl(
    const _Float16* __restrict__ A, const float* __restrict__ W,
    const float* __restrict__ bias, float scale,
    _Float16* __restrict__ outH, float* __restrict__ outF,
    const float* __restrict__ resid, int M, int Nout)
{
    constexpr int K = 128;
    __shared__ _Float16 As[128 * K];   // 32 KB: whole A block, read once from HBM
    __shared__ _Float16 Bs[64 * 32];   // 4 KB, stored [n][k] -> contiguous B frags
    int tid = threadIdx.x;
    int lane = tid & 31, w = tid >> 5;
    int hs = lane >> 4, l15 = lane & 15;
    int m0 = blockIdx.x * 128;
    int wm = w >> 1, wn = w & 1;       // 4 x 2 wave grid, 32x32 per wave

    { // stage full 128x128 A block once
        int row = tid >> 1, c0 = (tid & 1) << 6;
        const _Float16* src = A + (size_t)(m0 + row) * K + c0;
        _Float16* dst = As + row * K + c0;
#pragma unroll
        for (int j = 0; j < 64; ++j) dst[j] = src[j];
    }

    for (int nc = 0; nc < Nout; nc += 64) {
        v8f acc[2][2] = {};
        for (int k0 = 0; k0 < K; k0 += 32) {
            __syncthreads();           // covers A staging + prior Bs readers
            for (int idx = tid; idx < 64 * 32; idx += 256) {
                int n = idx >> 5, kk = idx & 31;
                Bs[n * 32 + kk] = (_Float16)W[(size_t)(nc + n) * K + k0 + kk];
            }
            __syncthreads();

            v16h a[2], b[2];
#pragma unroll
            for (int mt = 0; mt < 2; ++mt) {
                int row = wm * 32 + mt * 16 + l15;
#pragma unroll
                for (int e = 0; e < 16; ++e)
                    a[mt][e] = As[row * K + k0 + afrag_k(e, hs)];
            }
#pragma unroll
            for (int nt = 0; nt < 2; ++nt) {
                int col = wn * 32 + nt * 16 + l15;
#pragma unroll
                for (int e = 0; e < 16; ++e)
                    b[nt][e] = Bs[col * 32 + hs * 16 + e];
            }
#pragma unroll
            for (int mt = 0; mt < 2; ++mt)
#pragma unroll
                for (int nt = 0; nt < 2; ++nt)
                    acc[mt][nt] = __builtin_amdgcn_wmma_f32_16x16x32_f16(
                        false, a[mt], false, b[nt], (short)0, acc[mt][nt], false, false);
        }
        gemm_epilogue<MODE>(acc, m0, nc, wm, wn, hs, l15, bias, scale,
                            outH, outF, resid, Nout);
    }
}

// ======== WMMA GEMM, K-outer (for K=512 fc2): 128x64 tile per block ========
template<int MODE>
__global__ __launch_bounds__(256) void gemm_wmma(
    const _Float16* __restrict__ A, const float* __restrict__ W,
    const float* __restrict__ bias, float scale,
    _Float16* __restrict__ outH, float* __restrict__ outF,
    const float* __restrict__ resid, int M, int Nout, int K)
{
    __shared__ _Float16 As[128 * 32];   // 8 KB
    __shared__ _Float16 Bs[64 * 32];    // 4 KB, [n][k]
    int tid = threadIdx.x;
    int lane = tid & 31, w = tid >> 5;
    int hs = lane >> 4, l15 = lane & 15;
    int m0 = blockIdx.x * 128, n0 = blockIdx.y * 64;
    int wm = w >> 1, wn = w & 1;
    v8f acc[2][2] = {};

    for (int k0 = 0; k0 < K; k0 += 32) {
        {
            int row = tid >> 1, c0 = (tid & 1) << 4;
            const _Float16* src = A + (size_t)(m0 + row) * K + k0 + c0;
            if (k0 + 32 < K) __builtin_prefetch(src + 32, 0, 1);  // global_prefetch_b8
            _Float16* dst = As + row * 32 + c0;
#pragma unroll
            for (int j = 0; j < 16; ++j) dst[j] = src[j];
        }
        for (int idx = tid; idx < 64 * 32; idx += 256) {
            int n = idx >> 5, kk = idx & 31;
            Bs[n * 32 + kk] = (_Float16)W[(size_t)(n0 + n) * K + k0 + kk];
        }
        __syncthreads();

        v16h a[2], b[2];
#pragma unroll
        for (int mt = 0; mt < 2; ++mt) {
            int row = wm * 32 + mt * 16 + l15;
#pragma unroll
            for (int e = 0; e < 16; ++e) a[mt][e] = As[row * 32 + afrag_k(e, hs)];
        }
#pragma unroll
        for (int nt = 0; nt < 2; ++nt) {
            int col = wn * 32 + nt * 16 + l15;
#pragma unroll
            for (int e = 0; e < 16; ++e) b[nt][e] = Bs[col * 32 + hs * 16 + e];
        }
#pragma unroll
        for (int mt = 0; mt < 2; ++mt)
#pragma unroll
            for (int nt = 0; nt < 2; ++nt)
                acc[mt][nt] = __builtin_amdgcn_wmma_f32_16x16x32_f16(
                    false, a[mt], false, b[nt], (short)0, acc[mt][nt], false, false);
        __syncthreads();
    }
    gemm_epilogue<MODE>(acc, m0, n0, wm, wn, hs, l15, bias, scale,
                        outH, outF, resid, Nout);
}

// =================== Windowed attention: one WG per window, one wave/head ===
__global__ __launch_bounds__(128) void attn_kernel(
    const _Float16* __restrict__ Qg, const _Float16* __restrict__ Kg,
    const _Float16* __restrict__ Vg, const float* __restrict__ rpb,
    _Float16* __restrict__ Og)
{
    // 64 KB LDS: P[4][64x64] aliases Q (Q dead after scores)
    __shared__ _Float16 sU[16384];  // Q in first 8192 halves; P = sU + h*4096
    __shared__ _Float16 sK[8192];
    __shared__ _Float16 sV[8192];

    int tid = threadIdx.x;
    int win = blockIdx.x;
    size_t base = (size_t)win * 49 * CDIM;

    for (int i = tid; i < 64 * CDIM; i += 128) {
        int r = i >> 7;
        _Float16 q = (_Float16)0, k = (_Float16)0, v = (_Float16)0;
        if (r < 49) { q = Qg[base + i]; k = Kg[base + i]; v = Vg[base + i]; }
        sU[i] = q; sK[i] = k; sV[i] = v;
    }
    __syncthreads();

    int lane = tid & 31, h = tid >> 5;
    int hs = lane >> 4, l15 = lane & 15;

    // ---- S = (Q*scale) K^T : 4x4 tiles of 16x16, K-dim = 32 (head dim) ----
    v8f S[4][4];
#pragma unroll
    for (int mt = 0; mt < 4; ++mt) {
        v16h a;
        int row = mt * 16 + l15;
#pragma unroll
        for (int e = 0; e < 16; ++e)
            a[e] = sU[row * CDIM + h * 32 + afrag_k(e, hs)];
#pragma unroll
        for (int nt = 0; nt < 4; ++nt) {
            v16h b;
            int key = nt * 16 + l15;
#pragma unroll
            for (int e = 0; e < 16; ++e)
                b[e] = sK[key * CDIM + h * 32 + hs * 16 + e];
            v8f z = {};
            S[mt][nt] = __builtin_amdgcn_wmma_f32_16x16x32_f16(
                false, a, false, b, (short)0, z, false, false);
        }
    }

    // ---- relative position bias + padding mask ----
#pragma unroll
    for (int mt = 0; mt < 4; ++mt)
#pragma unroll
        for (int nt = 0; nt < 4; ++nt)
#pragma unroll
            for (int r = 0; r < 8; ++r) {
                int qi = mt * 16 + r + 8 * hs;
                int kj = nt * 16 + l15;
                if (qi < 49 && kj < 49) {
                    int yi = qi / 7, xi = qi % 7, yj = kj / 7, xj = kj % 7;
                    int idx = (yi - yj + 6) * 13 + (xi - xj + 6);
                    S[mt][nt][r] += rpb[idx * 4 + h];
                } else {
                    S[mt][nt][r] = -1e30f;
                }
            }

    __syncthreads();   // everyone done reading Q before P overwrites it
    _Float16* P = sU + h * 4096;

    // ---- softmax over rows (16-lane butterfly within C-layout halves) ----
#pragma unroll
    for (int mt = 0; mt < 4; ++mt)
#pragma unroll
        for (int r = 0; r < 8; ++r) {
            float mx = -1e30f;
#pragma unroll
            for (int nt = 0; nt < 4; ++nt) mx = fmaxf(mx, S[mt][nt][r]);
#pragma unroll
            for (int off = 1; off < 16; off <<= 1) mx = fmaxf(mx, __shfl_xor(mx, off, 32));
            float sum = 0.f;
#pragma unroll
            for (int nt = 0; nt < 4; ++nt) {
                float e2 = __expf(S[mt][nt][r] - mx);
                S[mt][nt][r] = e2; sum += e2;
            }
#pragma unroll
            for (int off = 1; off < 16; off <<= 1) sum += __shfl_xor(sum, off, 32);
            float inv = 1.0f / sum;
            int row = mt * 16 + r + 8 * hs;
#pragma unroll
            for (int nt = 0; nt < 4; ++nt)
                P[row * 64 + nt * 16 + l15] = (_Float16)(S[mt][nt][r] * inv);
        }

    // ---- O = P V : 4x2 tiles, K-dim = 64 (two 32-steps) ----
#pragma unroll
    for (int mt = 0; mt < 4; ++mt)
#pragma unroll
        for (int nt2 = 0; nt2 < 2; ++nt2) {
            v8f acc = {};
#pragma unroll
            for (int kt = 0; kt < 2; ++kt) {
                v16h a, b;
                int row = mt * 16 + l15;
#pragma unroll
                for (int e = 0; e < 16; ++e)
                    a[e] = P[row * 64 + kt * 32 + afrag_k(e, hs)];
                int col = h * 32 + nt2 * 16 + l15;
#pragma unroll
                for (int e = 0; e < 16; ++e)
                    b[e] = sV[(kt * 32 + hs * 16 + e) * CDIM + col];
                acc = __builtin_amdgcn_wmma_f32_16x16x32_f16(
                    false, a, false, b, (short)0, acc, false, false);
            }
#pragma unroll
            for (int r = 0; r < 8; ++r) {
                int qi = mt * 16 + r + 8 * hs;
                if (qi < 49)
                    Og[base + (size_t)qi * CDIM + h * 32 + nt2 * 16 + l15] = (_Float16)acc[r];
            }
        }
}

// ============================ host launcher ==============================
extern "C" void kernel_launch(void* const* d_in, const int* in_sizes, int n_in,
                              void* d_out, int out_size, void* d_ws, size_t ws_size,
                              hipStream_t stream) {
    const float* x      = (const float*)d_in[0];
    const float* v      = (const float*)d_in[1];
    const float* n1g    = (const float*)d_in[2];
    const float* n1b    = (const float*)d_in[3];
    const float* nvg    = (const float*)d_in[4];
    const float* nvb    = (const float*)d_in[5];
    const float* q_w    = (const float*)d_in[6];
    const float* q_b    = (const float*)d_in[7];
    const float* kv_w   = (const float*)d_in[8];
    const float* kv_b   = (const float*)d_in[9];
    const float* rpb    = (const float*)d_in[10];
    const float* proj_w = (const float*)d_in[11];
    const float* proj_b = (const float*)d_in[12];
    const float* n2g    = (const float*)d_in[13];
    const float* n2b    = (const float*)d_in[14];
    const float* fc1_w  = (const float*)d_in[15];
    const float* fc1_b  = (const float*)d_in[16];
    const float* fc2_w  = (const float*)d_in[17];
    const float* fc2_b  = (const float*)d_in[18];
    (void)in_sizes; (void)n_in; (void)out_size; (void)ws_size;

    const int M = MTOK;
    const size_t S1 = (size_t)M * CDIM * sizeof(_Float16);   // 51.4 MB
    char* ws = (char*)d_ws;
    // workspace plan with lifetime-based reuse (total = 8*S1 ~ 411 MB):
    _Float16* XN  = (_Float16*)(ws);              // LN(x), window order
    _Float16* VN  = (_Float16*)(ws + S1);         // LN(v), window order
    _Float16* Qb  = (_Float16*)(ws + 2 * S1);
    _Float16* Kb  = (_Float16*)(ws + 3 * S1);
    _Float16* Vb  = (_Float16*)(ws + 4 * S1);
    _Float16* AO  = XN;                           // attn out reuses XN
    float*    XO  = (float*)(ws + S1);            // f32, spans [S1, 3*S1)
    _Float16* H16 = (_Float16*)(ws + 3 * S1);     // reuses Kb
    _Float16* H1  = (_Float16*)(ws + 4 * S1);     // M x 512, spans [4*S1, 8*S1)
    float* out = (float*)d_out;

    // 1) LayerNorms, fused window-partition, f16 staging
    ln_kernel<true><<<M / 8, 256, 0, stream>>>(x, n1g, n1b, XN, M);
    ln_kernel<true><<<M / 8, 256, 0, stream>>>(v, nvg, nvb, VN, M);

    // 2) Q/K/V projections (WMMA, A staged once per block)
    gemm_wmma_nl<0><<<M / 128, 256, 0, stream>>>(
        XN, q_w, q_b, ATTN_SCALE, Qb, nullptr, nullptr, M, 128);
    gemm_wmma_nl<0><<<M / 128, 256, 0, stream>>>(
        VN, kv_w, kv_b, 1.0f, Kb, nullptr, nullptr, M, 128);
    gemm_wmma_nl<0><<<M / 128, 256, 0, stream>>>(
        VN, kv_w + 128 * 128, kv_b + 128, 1.0f, Vb, nullptr, nullptr, M, 128);

    // 3) windowed attention (softmax + rel-pos bias), WMMA for S and O
    attn_kernel<<<NWIN, 128, 0, stream>>>(Qb, Kb, Vb, rpb, AO);

    // 4) proj + window-reverse + residual -> xo (f32)
    gemm_wmma_nl<2><<<M / 128, 256, 0, stream>>>(
        AO, proj_w, proj_b, 1.0f, nullptr, XO, x, M, 128);

    // 5) LN2 + MLP: fc1+GELU reads activations once (N-loop), fc2 + residual
    ln_kernel<false><<<M / 8, 256, 0, stream>>>(XO, n2g, n2b, H16, M);
    gemm_wmma_nl<1><<<M / 128, 256, 0, stream>>>(
        H16, fc1_w, fc1_b, 1.0f, H1, nullptr, nullptr, M, 512);
    gemm_wmma<3><<<dim3(M / 128, 2), 256, 0, stream>>>(
        H1, fc2_w, fc2_b, 1.0f, nullptr, out, XO, M, 128, 512);
}
